// TimeAttention_1511828488344
// MI455X (gfx1250) — matrix operand order repaired and checked by reading, not compiled, not run on hardware
//
#include <hip/hip_runtime.h>

typedef __attribute__((ext_vector_type(16))) _Float16 v16h;
typedef __attribute__((ext_vector_type(8)))  _Float16 v8h;
typedef __attribute__((ext_vector_type(8)))  float    v8f;
typedef __attribute__((ext_vector_type(4)))  float    v4f;

constexpr int kB  = 4;
constexpr int kH  = 8;
constexpr int kD  = 64;
constexpr int kNV = 32;
constexpr int kNT = 64;
constexpr int kL  = kNV * kNT;   // 2048
constexpr int kHD = kH * kD;     // 512
// 1/sqrt(D) * log2(e): fold softmax scale + natural->base2 exp into Q and bias
constexpr float kScale = 0.125f * 1.4426950408889634f;

// LDS group-segment byte offset of a __shared__ object (generic pointers to
// LDS are {aperture_hi, 32-bit group offset}; truncation recovers the offset).
__device__ __forceinline__ uint32_t lds_off(const void* p) {
  return (uint32_t)(uintptr_t)p;
}

// 16-lane row reductions via DPP row_ror (VALU, no LDS traffic).
// row_ror:1 = 0x121, :2 = 0x122, :4 = 0x124, :8 = 0x128
template <int CTRL>
__device__ __forceinline__ float dppRot(float x) {
  return __int_as_float(__builtin_amdgcn_update_dpp(
      __float_as_int(x), __float_as_int(x), CTRL, 0xf, 0xf, false));
}
__device__ __forceinline__ float rowMax16(float x) {
  x = fmaxf(x, dppRot<0x121>(x));
  x = fmaxf(x, dppRot<0x122>(x));
  x = fmaxf(x, dppRot<0x124>(x));
  x = fmaxf(x, dppRot<0x128>(x));
  return x;
}
__device__ __forceinline__ float rowSum16(float x) {
  x += dppRot<0x121>(x);
  x += dppRot<0x122>(x);
  x += dppRot<0x124>(x);
  x += dppRot<0x128>(x);
  return x;
}

// ===========================================================================
// Pre-pass 1: KH[b][h][s][d] = (f16) K[b][s][h][d]
// ===========================================================================
__global__ __launch_bounds__(256)
void convert_k16(const float* __restrict__ K, _Float16* __restrict__ KH) {
  const int idx = blockIdx.x * 256 + threadIdx.x;   // B*H*L*8 = 524288 threads
  const int d0 = (idx & 7) * 8;
  const int s  = (idx >> 3) & (kL - 1);
  const int bh = idx >> 14;                         // L*8 = 16384 per (b,h)
  const int b  = bh >> 3, h = bh & 7;
  const v4f* src = (const v4f*)(K + (((size_t)b * kL + s) * kH + h) * kD + d0);
  v4f a = src[0], c = src[1];
  v8h o;
#pragma unroll
  for (int j = 0; j < 4; ++j) { o[j] = (_Float16)a[j]; o[4 + j] = (_Float16)c[j]; }
  *(v8h*)(KH + ((size_t)bh * kL + s) * kD + d0) = o;
}

// ===========================================================================
// Pre-pass 2: VT[b][h][d][s] = (f16) V[b][s][h][d]   (LDS tile transpose)
// ===========================================================================
__global__ __launch_bounds__(256)
void convert_v16t(const float* __restrict__ V, _Float16* __restrict__ VT) {
  __shared__ _Float16 tile[64][66];   // pad 66: conflict-free column reads
  const int bh = blockIdx.x >> 5;
  const int st = blockIdx.x & 31;     // 64-wide s tile
  const int b  = bh >> 3, h = bh & 7;
  const int t  = threadIdx.x;
  {
    const int s  = st * 64 + (t >> 2);
    const int d0 = (t & 3) * 16;
    const v4f* sv = (const v4f*)(V + (((size_t)b * kL + s) * kH + h) * kD + d0);
#pragma unroll
    for (int q = 0; q < 4; ++q) {
      v4f a = sv[q];
#pragma unroll
      for (int j = 0; j < 4; ++j) tile[t >> 2][d0 + 4 * q + j] = (_Float16)a[j];
    }
  }
  __syncthreads();
  {
    const int d  = t >> 2;
    const int s0 = (t & 3) * 16;
    v8h o0, o1;
#pragma unroll
    for (int j = 0; j < 8; ++j) {
      o0[j] = tile[s0 + j][d];
      o1[j] = tile[s0 + 8 + j][d];
    }
    _Float16* dst = VT + ((size_t)bh * kD + d) * kL + st * 64 + s0;
    *(v8h*)dst       = o0;
    *(v8h*)(dst + 8) = o1;
  }
}

// ===========================================================================
// Main attention kernel: async double-buffered LDS staging + WMMA f16
// ===========================================================================
__global__ __launch_bounds__(256)
void timeattn_fwd_async(const float* __restrict__ Q, const _Float16* __restrict__ KH,
                        const _Float16* __restrict__ VT, const float* __restrict__ BW,
                        float* __restrict__ Out)
{
  __shared__ __align__(32) _Float16 Kbuf[2][32 * 64];  // [key][d] f16
  __shared__ __align__(32) _Float16 Vbuf[2][64 * 32];  // [d][key] f16
  __shared__ __align__(32) _Float16 Plds[8][32 * 16];  // per-wave, col-major

  const int tid  = threadIdx.x;
  const int wv   = tid >> 5;
  const int lane = tid & 31;
  const int b    = blockIdx.x / kH;
  const int hh   = blockIdx.x % kH;
  const int sub  = blockIdx.y >> 2;     // query token block (16 tokens)
  const int vgrp = blockIdx.y & 3;
  const int v_q  = vgrp * 8 + wv;       // this wave's query variable
  const int qrow0 = v_q * kNT + sub * 16;

  const float w0s = BW[hh] * kScale;
  const float w1s = BW[kH + hh] * kScale;

  const int mrow = lane & 15;
  const int hlf  = lane >> 4;

  const int subp1 = sub + 1;
  const int kmul  = (subp1 == 1) ? 65536 : (subp1 == 2) ? 32768
                  : (subp1 == 3) ? 21846 : 16384;       // ceil(65536/subp1)
  const int nIter = 16 * subp1;

  // ---- Q tile (16x64) -> two WMMA f16 A operands, pre-scaled --------------
  v16h qa[2];
  {
    const v4f* qv = (const v4f*)(Q + ((size_t)b * kL + qrow0 + mrow) * kHD + hh * kD);
#pragma unroll
    for (int c = 0; c < 2; ++c) {
      v4f a0 = qv[(32 * c + 16 * hlf) >> 2];
      v4f a1 = qv[((32 * c + 16 * hlf) >> 2) + 1];
      v4f a2 = qv[(32 * c + 8 + 16 * hlf) >> 2];
      v4f a3 = qv[((32 * c + 8 + 16 * hlf) >> 2) + 1];
#pragma unroll
      for (int j = 0; j < 4; ++j) {
        qa[c][j]      = (_Float16)(a0[j] * kScale);
        qa[c][4 + j]  = (_Float16)(a1[j] * kScale);
        qa[c][8 + j]  = (_Float16)(a2[j] * kScale);
        qa[c][12 + j] = (_Float16)(a3[j] * kScale);
      }
    }
  }

  // ---- async staging setup ------------------------------------------------
  const uint64_t khB = (uint64_t)(KH + (size_t)(b * kH + hh) * (kL * kD));
  const uint64_t vtB = (uint64_t)(VT + (size_t)(b * kH + hh) * (kL * kD));
  const int      kHalf  = tid >> 7;          // which 16-key block of the pair
  const uint32_t kOffIn = (tid & 127) * 16;  // bytes inside 2KB block
  const int      vD  = tid & 63;
  const int      vQ2 = tid >> 6;             // 0..3
  const uint32_t kLds0 = lds_off(&Kbuf[0][0]);
  const uint32_t vLds0 = lds_off(&Vbuf[0][0]);

  auto prefetch = [&](int it2, int bufSel) {
    {  // K: two 16-key blocks, each 2KB contiguous, 1 x b128 per thread
      const int blk = 2 * it2 + kHalf;
      const int vk  = (blk * kmul) >> 16;
      const int tkb = blk - vk * subp1;
      const uint32_t goff = (uint32_t)((vk * kNT + tkb * 16) * (kD * 2)) + kOffIn;
      const uint32_t la   = kLds0 + bufSel * 4096 + kHalf * 2048 + kOffIn;
      asm volatile("global_load_async_to_lds_b128 %0, %1, %2"
                   :: "v"(la), "v"(goff), "s"(khB) : "memory");
    }
    {  // V^T: per d-row, two 16-key segments; 1 x b128 per thread
      const int blk = 2 * it2 + (vQ2 >> 1);
      const int vk  = (blk * kmul) >> 16;
      const int tkb = blk - vk * subp1;
      const int s0  = vk * kNT + tkb * 16 + 8 * (vQ2 & 1);
      const uint32_t goff = (uint32_t)((vD * kL + s0) * 2);
      const uint32_t la   = vLds0 + bufSel * 4096 +
                            (vD * 32 + 16 * (vQ2 >> 1) + 8 * (vQ2 & 1)) * 2;
      asm volatile("global_load_async_to_lds_b128 %0, %1, %2"
                   :: "v"(la), "v"(goff), "s"(vtB) : "memory");
    }
  };

  // ---- softmax state ------------------------------------------------------
  float mrun[8], lrun[8];
  v8f oacc[4];
#pragma unroll
  for (int r = 0; r < 8; ++r) { mrun[r] = -__builtin_inff(); lrun[r] = 0.0f; }
#pragma unroll
  for (int nc = 0; nc < 4; ++nc)
#pragma unroll
    for (int r = 0; r < 8; ++r) oacc[nc][r] = 0.0f;

  prefetch(0, 0);

  for (int it = 0; it < nIter; ++it) {
    const int cur = it & 1;
    asm volatile("s_wait_asynccnt 0x0" ::: "memory");
    __syncthreads();
    if (it + 1 < nIter) prefetch(it + 1, cur ^ 1);

    const _Float16* Kc = &Kbuf[cur][0];
    const _Float16* Vc = &Vbuf[cur][0];

    // ---- S = Q*K^T : two 16x16 tiles --------------------------------------
    float st[2][8];
#pragma unroll
    for (int t = 0; t < 2; ++t) {
      v8f acc;
#pragma unroll
      for (int r = 0; r < 8; ++r) acc[r] = 0.0f;
#pragma unroll
      for (int c = 0; c < 2; ++c) {
        v16h bk = *(const v16h*)&Kc[(16 * t + mrow) * 64 + 32 * c + 16 * hlf];
        acc = __builtin_amdgcn_wmma_f32_16x16x32_f16(false, qa[c], false, bk,
                                                     (short)0, acc, false, false);
      }
      const int blk = 2 * it + t;
      const int vk  = (blk * kmul) >> 16;
      const int tkb = blk - vk * subp1;
      const float bias = (vk == v_q) ? w1s : w0s;
#pragma unroll
      for (int r = 0; r < 8; ++r) st[t][r] = acc[r] + bias;
      if (tkb == sub) {  // wave-uniform: only diagonal token block is masked
#pragma unroll
        for (int r = 0; r < 8; ++r)
          st[t][r] = (mrow > r + 8 * hlf) ? -__builtin_inff() : st[t][r];
      }
    }

    // ---- online softmax: row max via DPP rotations ------------------------
    float rmax[8];
#pragma unroll
    for (int r = 0; r < 8; ++r)
      rmax[r] = rowMax16(fmaxf(st[0][r], st[1][r]));

    // Rescale only when some running max actually increased (wave-uniform).
    int upd = 0;
#pragma unroll
    for (int r = 0; r < 8; ++r) upd |= (rmax[r] > mrun[r]) ? 1 : 0;
    if (__any(upd)) {
#pragma unroll
      for (int r = 0; r < 8; ++r) {
        const float mn    = fmaxf(mrun[r], rmax[r]);
        const float alpha = __builtin_exp2f(mrun[r] - mn);
        mrun[r] = mn;
        lrun[r] *= alpha;
#pragma unroll
        for (int nc = 0; nc < 4; ++nc) oacc[nc][r] *= alpha;
      }
    }

    float p0[8], p1[8], rsum[8];
#pragma unroll
    for (int r = 0; r < 8; ++r) {
      p0[r]   = __builtin_exp2f(st[0][r] - mrun[r]);
      p1[r]   = __builtin_exp2f(st[1][r] - mrun[r]);
      rsum[r] = rowSum16(p0[r] + p1[r]);
      lrun[r] += rsum[r];
    }

    // ---- P to f16 via per-wave LDS (store col-major, reload A-layout) -----
    _Float16* Pb = &Plds[wv][0];
    {
      v8h ph0, ph1;
#pragma unroll
      for (int r = 0; r < 8; ++r) {
        ph0[r] = (_Float16)p0[r];
        ph1[r] = (_Float16)p1[r];
      }
      *(v8h*)&Pb[mrow * 16 + 8 * hlf]        = ph0;
      *(v8h*)&Pb[(16 + mrow) * 16 + 8 * hlf] = ph1;
    }
    asm volatile("s_wait_dscnt 0" ::: "memory");
    v16h pa;
#pragma unroll
    for (int j = 0; j < 8; ++j) {
      pa[j]     = Pb[(16 * hlf + j) * 16 + mrow];
      pa[8 + j] = Pb[(16 * hlf + 8 + j) * 16 + mrow];
    }

    // ---- O += P * V -------------------------------------------------------
#pragma unroll
    for (int nc = 0; nc < 4; ++nc) {
      v16h bv = *(const v16h*)&Vc[(16 * nc + mrow) * 32 + 16 * hlf];
      oacc[nc] = __builtin_amdgcn_wmma_f32_16x16x32_f16(false, pa, false, bv,
                                                        (short)0, oacc[nc],
                                                        false, false);
    }
  }

  // ---- epilogue: O / l ----------------------------------------------------
  float* op = Out + ((size_t)b * kL + qrow0 + 8 * hlf) * kHD + hh * kD + mrow;
#pragma unroll
  for (int r = 0; r < 8; ++r) {
    const float inv = 1.0f / lrun[r];
#pragma unroll
    for (int nc = 0; nc < 4; ++nc)
      op[r * kHD + 16 * nc] = oacc[nc][r] * inv;
  }
}

// ===========================================================================
// Fallback (no workspace): in-register staging variant
// ===========================================================================
__global__ __launch_bounds__(256)
void timeattn_fwd_reg(const float* __restrict__ Q, const float* __restrict__ K,
                      const float* __restrict__ V, const float* __restrict__ BW,
                      float* __restrict__ Out)
{
  __shared__ __align__(32) _Float16 Klds[32 * 64];
  __shared__ __align__(32) _Float16 Vtlds[64 * 32];
  __shared__ __align__(32) _Float16 Plds[8][32 * 16];

  const int tid  = threadIdx.x;
  const int wv   = tid >> 5;
  const int lane = tid & 31;
  const int b    = blockIdx.x / kH;
  const int hh   = blockIdx.x % kH;
  const int sub  = blockIdx.y >> 2;
  const int vgrp = blockIdx.y & 3;
  const int v_q  = vgrp * 8 + wv;
  const int qrow0 = v_q * kNT + sub * 16;

  const float w0s = BW[hh] * kScale;
  const float w1s = BW[kH + hh] * kScale;

  const int mrow = lane & 15;
  const int hlf  = lane >> 4;

  v16h qa[2];
  {
    const v4f* qv = (const v4f*)(Q + ((size_t)b * kL + qrow0 + mrow) * kHD + hh * kD);
#pragma unroll
    for (int c = 0; c < 2; ++c) {
      v4f a0 = qv[(32 * c + 16 * hlf) >> 2];
      v4f a1 = qv[((32 * c + 16 * hlf) >> 2) + 1];
      v4f a2 = qv[(32 * c + 8 + 16 * hlf) >> 2];
      v4f a3 = qv[((32 * c + 8 + 16 * hlf) >> 2) + 1];
#pragma unroll
      for (int j = 0; j < 4; ++j) {
        qa[c][j]      = (_Float16)(a0[j] * kScale);
        qa[c][4 + j]  = (_Float16)(a1[j] * kScale);
        qa[c][8 + j]  = (_Float16)(a2[j] * kScale);
        qa[c][12 + j] = (_Float16)(a3[j] * kScale);
      }
    }
  }

  const int kp  = tid >> 3;
  const int kd0 = (tid & 7) * 8;
  const int vd  = tid & 63;
  const int vc  = tid >> 6;

  float mrun[8], lrun[8];
  v8f oacc[4];
#pragma unroll
  for (int r = 0; r < 8; ++r) { mrun[r] = -__builtin_inff(); lrun[r] = 0.0f; }
#pragma unroll
  for (int nc = 0; nc < 4; ++nc)
#pragma unroll
    for (int r = 0; r < 8; ++r) oacc[nc][r] = 0.0f;

  const int subp1 = sub + 1;
  const int nIter = 16 * subp1;

  for (int it = 0; it < nIter; ++it) {
    __syncthreads();
    {
      const int blk  = 2 * it + (kp >> 4);
      const int vk   = blk / subp1;
      const int tkb  = blk % subp1;
      const int srow = vk * kNT + tkb * 16 + (kp & 15);
      const v4f* kg = (const v4f*)(K + ((size_t)b * kL + srow) * kHD + hh * kD + kd0);
      v4f k0 = kg[0], k1 = kg[1];
      v8h khf;
#pragma unroll
      for (int j = 0; j < 4; ++j) { khf[j] = (_Float16)k0[j]; khf[4 + j] = (_Float16)k1[j]; }
      *(v8h*)&Klds[kp * 64 + kd0] = khf;
    }
    {
      const int blk = 2 * it + (vc >> 1);
      const int vk  = blk / subp1;
      const int tkb = blk % subp1;
      v8h vhf;
#pragma unroll
      for (int j = 0; j < 8; ++j) {
        const int slot = 8 * vc + j;
        const int srow = vk * kNT + tkb * 16 + (slot & 15);
        vhf[j] = (_Float16)V[((size_t)b * kL + srow) * kHD + hh * kD + vd];
      }
      *(v8h*)&Vtlds[vd * 32 + 8 * vc] = vhf;
    }
    __syncthreads();

    float st[2][8];
#pragma unroll
    for (int t = 0; t < 2; ++t) {
      v8f acc;
#pragma unroll
      for (int r = 0; r < 8; ++r) acc[r] = 0.0f;
#pragma unroll
      for (int c = 0; c < 2; ++c) {
        v16h bk = *(const v16h*)&Klds[(16 * t + mrow) * 64 + 32 * c + 16 * hlf];
        acc = __builtin_amdgcn_wmma_f32_16x16x32_f16(false, qa[c], false, bk,
                                                     (short)0, acc, false, false);
      }
      const int blk = 2 * it + t;
      const int vk  = blk / subp1;
      const int tkb = blk % subp1;
      const float bias = (vk == v_q) ? w1s : w0s;
#pragma unroll
      for (int r = 0; r < 8; ++r) st[t][r] = acc[r] + bias;
      if (tkb == sub) {
#pragma unroll
        for (int r = 0; r < 8; ++r)
          st[t][r] = (mrow > r + 8 * hlf) ? -__builtin_inff() : st[t][r];
      }
    }

    float rmax[8];
#pragma unroll
    for (int r = 0; r < 8; ++r)
      rmax[r] = rowMax16(fmaxf(st[0][r], st[1][r]));

    int upd = 0;
#pragma unroll
    for (int r = 0; r < 8; ++r) upd |= (rmax[r] > mrun[r]) ? 1 : 0;
    if (__any(upd)) {
#pragma unroll
      for (int r = 0; r < 8; ++r) {
        const float mn    = fmaxf(mrun[r], rmax[r]);
        const float alpha = __builtin_exp2f(mrun[r] - mn);
        mrun[r] = mn;
        lrun[r] *= alpha;
#pragma unroll
        for (int nc = 0; nc < 4; ++nc) oacc[nc][r] *= alpha;
      }
    }

    float p0[8], p1[8];
#pragma unroll
    for (int r = 0; r < 8; ++r) {
      p0[r] = __builtin_exp2f(st[0][r] - mrun[r]);
      p1[r] = __builtin_exp2f(st[1][r] - mrun[r]);
      lrun[r] += rowSum16(p0[r] + p1[r]);
    }

    _Float16* Pb = &Plds[wv][0];
    {
      v8h ph0, ph1;
#pragma unroll
      for (int r = 0; r < 8; ++r) { ph0[r] = (_Float16)p0[r]; ph1[r] = (_Float16)p1[r]; }
      *(v8h*)&Pb[mrow * 16 + 8 * hlf]        = ph0;
      *(v8h*)&Pb[(16 + mrow) * 16 + 8 * hlf] = ph1;
    }
    asm volatile("s_wait_dscnt 0" ::: "memory");
    v16h pa;
#pragma unroll
    for (int j = 0; j < 8; ++j) {
      pa[j]     = Pb[(16 * hlf + j) * 16 + mrow];
      pa[8 + j] = Pb[(16 * hlf + 8 + j) * 16 + mrow];
    }

#pragma unroll
    for (int nc = 0; nc < 4; ++nc) {
      v16h bv = *(const v16h*)&Vtlds[(16 * nc + mrow) * 32 + 16 * hlf];
      oacc[nc] = __builtin_amdgcn_wmma_f32_16x16x32_f16(false, pa, false, bv,
                                                        (short)0, oacc[nc],
                                                        false, false);
    }
  }

  float* op = Out + ((size_t)b * kL + qrow0 + 8 * hlf) * kHD + hh * kD + mrow;
#pragma unroll
  for (int r = 0; r < 8; ++r) {
    const float inv = 1.0f / lrun[r];
#pragma unroll
    for (int nc = 0; nc < 4; ++nc)
      op[r * kHD + 16 * nc] = oacc[nc][r] * inv;
  }
}

// ===========================================================================
extern "C" void kernel_launch(void* const* d_in, const int* in_sizes, int n_in,
                              void* d_out, int out_size, void* d_ws, size_t ws_size,
                              hipStream_t stream) {
  (void)in_sizes; (void)n_in; (void)out_size;
  const float* Q  = (const float*)d_in[0];
  const float* K  = (const float*)d_in[1];
  const float* V  = (const float*)d_in[2];
  const float* BW = (const float*)d_in[3];
  float* Out = (float*)d_out;

  const size_t perMat = (size_t)kB * kH * kL * kD;        // 4M halfs = 8MB
  const size_t need   = 2 * perMat * sizeof(_Float16);    // 16MB

  dim3 grid(kB * kH, 16);
  if (ws_size >= need) {
    _Float16* KH = (_Float16*)d_ws;
    _Float16* VT = KH + perMat;
    convert_k16 <<<(kB * kH * kL * 8) / 256, 256, 0, stream>>>(K, KH);
    convert_v16t<<<kB * kH * (kL / 64),     256, 0, stream>>>(V, VT);
    timeattn_fwd_async<<<grid, 256, 0, stream>>>(Q, KH, VT, BW, Out);
  } else {
    timeattn_fwd_reg<<<grid, 256, 0, stream>>>(Q, K, V, BW, Out);
  }
}
